// ImbalancedTripletContrastiveLoss_36258113912946
// MI455X (gfx1250) — matrix operand order repaired and checked by reading, not compiled, not run on hardware
//
#include <hip/hip_runtime.h>

typedef __attribute__((ext_vector_type(16))) _Float16 v16h;
typedef __attribute__((ext_vector_type(8)))  float    v8f;

#define B_N 4096
#define T_N 128
#define D_N 64
#define E_N 128

__constant__ const float kTEMP = 0.1f;
__constant__ const float kBASE_TEMP = 0.07f;
__constant__ const float kMARGIN = 1.0f;
__constant__ const float kLAMBDA_C = 5.0f;
__constant__ const float kLAMBDA_T = 2.0f;

// ---- ws layout (float offsets) ----
#define WS_MSE    0
#define WS_CNT    1   // [2]
#define WS_OWN    3   // [2]
#define WS_REP    5   // [2]
#define WS_TRIP   7
#define WS_CON    8
#define WS_CENTER 16  // [2][128]
#define WS_EMB32  512 // [4096][128] f32, then f16 copy after

// ---------------- helpers ----------------
__device__ inline float wave_sum32(float v) {
  #pragma unroll
  for (int o = 16; o > 0; o >>= 1) v += __shfl_down(v, o, 32);
  return v;
}

// ---------------- K0: zero accumulators ----------------
__global__ void init_ws_kernel(float* ws) {
  int t = blockIdx.x * blockDim.x + threadIdx.x;
  if (t < WS_CENTER + 2 * E_N) ws[t] = 0.0f;
}

// ---------------- K1: MSE partial sums ----------------
__global__ __launch_bounds__(256)
void mse_kernel(const float4* __restrict__ rec, const float4* __restrict__ inp,
                float* __restrict__ acc, int n4) {
  float s = 0.0f;
  int stride = gridDim.x * blockDim.x;
  for (int i = blockIdx.x * blockDim.x + threadIdx.x; i < n4; i += stride) {
    float4 a = rec[i], b = inp[i];
    float dx = a.x - b.x, dy = a.y - b.y, dz = a.z - b.z, dw = a.w - b.w;
    s += dx * dx + dy * dy + dz * dz + dw * dw;
  }
  __shared__ float red[8];
  s = wave_sum32(s);
  if ((threadIdx.x & 31) == 0) red[threadIdx.x >> 5] = s;
  __syncthreads();
  if (threadIdx.x == 0) {
    float t = 0.0f;
    #pragma unroll
    for (int w = 0; w < 8; w++) t += red[w];
    atomicAdd(acc, t);
  }
}

// ---------------- K2: max-pool over T, L2 normalize, center accum ----------------
__global__ __launch_bounds__(128)
void embed_kernel(const float* __restrict__ enc, const int* __restrict__ labels,
                  float* __restrict__ emb32, _Float16* __restrict__ emb16,
                  float* __restrict__ ws) {
  int b = blockIdx.x;
  int e = threadIdx.x;
  const float* p = enc + (size_t)b * T_N * E_N + e;
  float mx = -3.4e38f;
  #pragma unroll 4
  for (int t = 0; t < T_N; t++) mx = fmaxf(mx, p[(size_t)t * E_N]);

  __shared__ float red[4];
  float ss = wave_sum32(mx * mx);
  if ((threadIdx.x & 31) == 0) red[threadIdx.x >> 5] = ss;
  __syncthreads();
  float tot = red[0] + red[1] + red[2] + red[3];
  float denom = fmaxf(sqrtf(tot), 1e-12f);
  float v = mx / denom;

  emb32[b * E_N + e] = v;
  emb16[b * E_N + e] = (_Float16)v;

  int lbl = labels[b];
  atomicAdd(&ws[WS_CENTER + lbl * E_N + e], v);
  if (e == 0) atomicAdd(&ws[WS_CNT + lbl], 1.0f);
}

// ---------------- K3: finalize centers ----------------
__global__ void center_fin_kernel(float* ws) {
  int t = threadIdx.x;          // 256 threads
  int c = t >> 7, e = t & 127;
  ws[WS_CENTER + c * E_N + e] /= ws[WS_CNT + c];
}

// ---------------- K4: cluster loss partials ----------------
__global__ __launch_bounds__(128)
void cluster_kernel(const float* __restrict__ emb32, const int* __restrict__ labels,
                    float* __restrict__ ws) {
  int b = blockIdx.x, e = threadIdx.x;
  const float* centers = ws + WS_CENTER;
  float v = emb32[b * E_N + e];
  float d0 = v - centers[e];
  float d1 = v - centers[E_N + e];
  __shared__ float r0[4], r1[4];
  float s0 = wave_sum32(d0 * d0);
  float s1 = wave_sum32(d1 * d1);
  if ((threadIdx.x & 31) == 0) { r0[threadIdx.x >> 5] = s0; r1[threadIdx.x >> 5] = s1; }
  __syncthreads();
  if (threadIdx.x == 0) {
    float t0 = r0[0] + r0[1] + r0[2] + r0[3];
    float t1 = r1[0] + r1[1] + r1[2] + r1[3];
    float dist0 = sqrtf(t0), dist1 = sqrtf(t1);
    int l = labels[b];
    float own = l ? dist1 : dist0;
    float oth = l ? dist0 : dist1;
    atomicAdd(&ws[WS_OWN + l], own);
    atomicAdd(&ws[WS_REP + l], __expf(-oth));
  }
}

// ---------------- K5: triplet loss partials ----------------
__global__ __launch_bounds__(128)
void triplet_kernel(const float* __restrict__ emb32, const int* __restrict__ labels,
                    const int* __restrict__ ta, const int* __restrict__ tp,
                    const int* __restrict__ tn, float* __restrict__ ws, int ntri) {
  int t = blockIdx.x;
  if (t >= ntri) return;
  int e = threadIdx.x;
  int ia = ta[t], ip = tp[t], iq = tn[t];
  float a = emb32[ia * E_N + e];
  float p = emb32[ip * E_N + e];
  float q = emb32[iq * E_N + e];
  float dp = (a - p) * (a - p);
  float dn = (a - q) * (a - q);
  __shared__ float r0[4], r1[4];
  float s0 = wave_sum32(dp);
  float s1 = wave_sum32(dn);
  if ((threadIdx.x & 31) == 0) { r0[threadIdx.x >> 5] = s0; r1[threadIdx.x >> 5] = s1; }
  __syncthreads();
  if (threadIdx.x == 0) {
    float pd = r0[0] + r0[1] + r0[2] + r0[3];
    float nd = r1[0] + r1[1] + r1[2] + r1[3];
    float l = fmaxf(pd - nd + kMARGIN, 0.0f);
    float w = (labels[ia] == 1) ? 2.0f : 1.0f;
    atomicAdd(&ws[WS_TRIP], w * l);
  }
}

// ---------------- K6: fused WMMA Gram + fixed-shift-softmax SupCon ----------------
// A fragment (16x32 MxK f16): lane<16 -> M=lane, halfs [K0..7, K16..23];
//                             lane>=16 -> M=lane-16, halfs [K8..15, K24..31].
__device__ inline v16h load_a_frag(const _Float16* emb, int m0, int k0, int lane) {
  int m  = m0 + (lane & 15);
  int kb = k0 + ((lane >> 4) << 3);
  const _Float16* p = emb + m * E_N + kb;
  union { v16h v; float4 f[2]; } u;
  u.f[0] = *(const float4*)(p);
  u.f[1] = *(const float4*)(p + 16);
  return u.v;
}
// B fragment (32x16 KxN f16), B[k][n] = emb[n0+n][k0+k]:
// lane<16 -> N=lane, halfs [K0..7, K16..23]; lane>=16 -> N=lane-16, halfs [K8..15, K24..31].
// Two 16-byte loads per lane from row-major emb (no transpose needed).
__device__ inline v16h load_b_frag(const _Float16* emb, int n0, int k0, int lane) {
  int n  = n0 + (lane & 15);
  int kb = k0 + ((lane >> 4) << 3);
  const _Float16* p = emb + n * E_N + kb;
  union { v16h v; float4 f[2]; } u;
  u.f[0] = *(const float4*)(p);
  u.f[1] = *(const float4*)(p + 16);
  return u.v;
}

// 512 wave-jobs = 256 M-tiles x 2 column halves; pairs of waves in a block
// share an M-tile and merge plain (max, sum, sum) partials through LDS.
__global__ __launch_bounds__(256)
void supcon_kernel(const _Float16* __restrict__ emb16, const int* __restrict__ labels,
                   const float* __restrict__ ws, float* __restrict__ con_acc) {
  int lane = threadIdx.x & 31;
  int wave = threadIdx.x >> 5;
  int gp   = blockIdx.x * 8 + wave;   // 0..511
  int mt   = gp >> 1;                 // M-tile 0..255
  int ch   = gp & 1;                  // column half
  int m0   = mt * 16;
  int half = lane >> 4;
  int col  = lane & 15;

  // A fragments for all of K=128 (4 k-blocks of 32), kept in registers
  v16h a0 = load_a_frag(emb16, m0, 0,  lane);
  v16h a1 = load_a_frag(emb16, m0, 32, lane);
  v16h a2 = load_a_frag(emb16, m0, 64, lane);
  v16h a3 = load_a_frag(emb16, m0, 96, lane);

  int lab_i[8];
  float flab_i[8];
  #pragma unroll
  for (int r = 0; r < 8; r++) {
    lab_i[r]  = labels[m0 + r + 8 * half];  // C/D layout: VGPR r -> row m0+r (+8 upper half)
    flab_i[r] = (float)lab_i[r];
  }

  // Fixed-shift softmax: z = dot/TEMP bounded by 1/TEMP = 10 (normalized embs),
  // accumulate exp(z - 10) (no over/underflow), true row max tracked by plain fmax.
  // P via two unconditional sums: Pa = sum z (all j), P1 = sum labels_j * z;
  // diagonal handled once (wave-uniform n0 == m0 tile) via recorded corrections.
  const float invT  = 1.0f / kTEMP;
  const float SHIFT = invT;

  float m[8], S[8], Pa[8], P1[8], zd[8], ed[8];
  #pragma unroll
  for (int r = 0; r < 8; r++) {
    m[r] = -3.0e38f; S[r] = 0.0f; Pa[r] = 0.0f; P1[r] = 0.0f;
    zd[r] = 0.0f; ed[r] = 0.0f;
  }

  int nbeg = ch * (B_N / 2);
  int nend = nbeg + (B_N / 2);
  for (int n0 = nbeg; n0 < nend; n0 += 16) {
    v8f acc = {0.f, 0.f, 0.f, 0.f, 0.f, 0.f, 0.f, 0.f};
    acc = __builtin_amdgcn_wmma_f32_16x16x32_f16(false, a0, false,
            load_b_frag(emb16, n0, 0,  lane), (short)0, acc, false, false);
    acc = __builtin_amdgcn_wmma_f32_16x16x32_f16(false, a1, false,
            load_b_frag(emb16, n0, 32, lane), (short)0, acc, false, false);
    acc = __builtin_amdgcn_wmma_f32_16x16x32_f16(false, a2, false,
            load_b_frag(emb16, n0, 64, lane), (short)0, acc, false, false);
    acc = __builtin_amdgcn_wmma_f32_16x16x32_f16(false, a3, false,
            load_b_frag(emb16, n0, 96, lane), (short)0, acc, false, false);

    float fl = (float)labels[n0 + col];   // this lane's column label, all 8 rows
    #pragma unroll
    for (int r = 0; r < 8; r++) {
      float z  = acc[r] * invT;
      m[r]  = fmaxf(m[r], z);             // row max INCLUDES diagonal (matches jnp.max)
      float ez = __expf(z - SHIFT);
      S[r]  += ez;
      Pa[r] += z;
      P1[r] = fmaf(fl, z, P1[r]);
    }

    if (n0 == m0) {                       // wave-uniform scalar branch: diagonal tile
      #pragma unroll
      for (int r = 0; r < 8; r++) {
        bool d = (col == r + 8 * half);   // the one lane holding z_ii for this row
        float z  = acc[r] * invT;
        float ez = __expf(z - SHIFT);
        zd[r] = d ? z  : 0.0f;
        ed[r] = d ? ez : 0.0f;
      }
    }
  }

  // apply diagonal corrections, resolve P, then max/add/add butterfly per half-wave
  float cnt[2] = { ws[WS_CNT + 0], ws[WS_CNT + 1] };

  __shared__ float sm[8][16], sS[8][16], sP[8][16];
  #pragma unroll
  for (int r = 0; r < 8; r++) {
    float SS = S[r]  - ed[r];                       // drop exp(z_ii - SHIFT)
    float PA = Pa[r] - zd[r];                       // drop z_ii
    float Pone = P1[r] - flab_i[r] * zd[r];         // diag label == row label
    float PP = lab_i[r] ? Pone : (PA - Pone);       // sum over same-label off-diag
    float mm = m[r];
    #pragma unroll
    for (int o = 1; o < 16; o <<= 1) {
      mm = fmaxf(mm, __shfl_xor(mm, o, 32));
      SS += __shfl_xor(SS, o, 32);
      PP += __shfl_xor(PP, o, 32);
    }
    if (col == 0) {                                 // lanes 0 / 16 hold 8 rows each
      int row = r + 8 * half;
      sm[wave][row] = mm; sS[wave][row] = SS; sP[wave][row] = PP;
    }
  }
  __syncthreads();

  // even wave of each pair merges the two column halves and finalizes 16 rows
  if (((wave & 1) == 0) && lane < 16) {
    int row = lane;
    float mm = fmaxf(sm[wave][row], sm[wave + 1][row]);
    float SS = sS[wave][row] + sS[wave + 1][row];
    float PP = sP[wave][row] + sP[wave + 1][row];
    int   li = labels[m0 + row];
    float npos = cnt[li] - 1.0f;
    float Sref = SS * __expf(SHIFT - mm);           // Σ_{j≠i} exp(z_ij - m_i)
    float mlp  = (PP - npos * mm - npos * __logf(Sref + 1e-8f)) / (npos + 1e-8f);
    float c    = -(kTEMP / kBASE_TEMP) * mlp;
    #pragma unroll
    for (int o = 1; o < 16; o <<= 1) c += __shfl_xor(c, o, 32);
    if (lane == 0) atomicAdd(con_acc, c);
  }
}

// ---------------- K7: combine everything ----------------
__global__ void finalize_kernel(const float* __restrict__ ws, float* __restrict__ out,
                                int ntri) {
  float recon = ws[WS_MSE] / (float)((size_t)B_N * T_N * D_N);
  float n = (float)B_N;
  float cl = 0.0f;
  #pragma unroll
  for (int c = 0; c < 2; c++) {
    float cnt = ws[WS_CNT + c];
    float w = (n - cnt) / n;          // w.sum()==1 since counts sum to n
    cl += w * (ws[WS_OWN + c] / cnt) + w * (ws[WS_REP + c] / cnt);
  }
  cl *= 0.5f;
  float trip  = ws[WS_TRIP] / (float)ntri;
  float cmean = ws[WS_CON] / n;
  float contra = cmean + cl + kLAMBDA_T * trip;
  float total  = recon + kLAMBDA_C * contra;
  out[0] = total;
  out[1] = recon;
  out[2] = contra;
}

extern "C" void kernel_launch(void* const* d_in, const int* in_sizes, int n_in,
                              void* d_out, int out_size, void* d_ws, size_t ws_size,
                              hipStream_t stream) {
  const float* rec    = (const float*)d_in[0];
  const float* inp    = (const float*)d_in[1];
  const float* enc    = (const float*)d_in[2];
  const int*   labels = (const int*)d_in[3];
  const int*   ta     = (const int*)d_in[4];
  const int*   tp     = (const int*)d_in[5];
  const int*   tn     = (const int*)d_in[6];
  int ntri = in_sizes[4];

  float*     ws    = (float*)d_ws;
  float*     emb32 = ws + WS_EMB32;
  _Float16*  emb16 = (_Float16*)(emb32 + (size_t)B_N * E_N);

  init_ws_kernel<<<1, 512, 0, stream>>>(ws);

  int n4 = (B_N * T_N * D_N) / 4;  // 8,388,608 float4 elements
  mse_kernel<<<2048, 256, 0, stream>>>((const float4*)rec, (const float4*)inp,
                                       ws + WS_MSE, n4);

  embed_kernel<<<B_N, 128, 0, stream>>>(enc, labels, emb32, emb16, ws);
  center_fin_kernel<<<1, 256, 0, stream>>>(ws);
  cluster_kernel<<<B_N, 128, 0, stream>>>(emb32, labels, ws);
  triplet_kernel<<<ntri, 128, 0, stream>>>(emb32, labels, ta, tp, tn, ws, ntri);

  supcon_kernel<<<64, 256, 0, stream>>>(emb16, labels, ws, ws + WS_CON);

  finalize_kernel<<<1, 1, 0, stream>>>(ws, (float*)d_out, ntri);
}
